// FullAttnRes_67757404062020
// MI455X (gfx1250) — matrix operand order repaired and checked by reading, not compiled, not run on hardware
//
#include <hip/hip_runtime.h>
#include <hip/hip_bf16.h>

// ---------------------------------------------------------------------------
// Types for CDNA5 WMMA (wave32): 16x16x32 bf16 -> f32
// ---------------------------------------------------------------------------
typedef __attribute__((ext_vector_type(16))) __bf16 v16bf;
typedef __attribute__((ext_vector_type(8)))  float  v8f;

union FragBF16 {
    uint4 u[2];   // 32 bytes
    v16bf v;      // 16 x bf16
};

#define BT      16384            // B*T = 8*2048
#define DIM     512
#define NLAYERS 8

// ---------------------------------------------------------------------------
// prep_w: layer_weights f32 [L][D(k)][D(n)]  ->  Wt bf16 [L][N][K]  (N-major)
// ---------------------------------------------------------------------------
__global__ __launch_bounds__(256)
void prep_w_kernel(const float* __restrict__ lw, __hip_bfloat16* __restrict__ wt) {
    size_t o   = (size_t)blockIdx.x * 256 + threadIdx.x;   // 0 .. 8*512*512-1
    size_t l   = o >> 18;                                  // / (512*512)
    size_t rem = o & ((1u << 18) - 1);
    size_t n   = rem >> 9;                                 // / 512
    size_t k   = rem & 511;
    wt[o] = __float2bfloat16(lw[(l << 18) + (k << 9) + n]);
}

// ---------------------------------------------------------------------------
// mix_kernel: one block per (b,t) row.  Fused RMSNorm-dot + softmax + mixture.
//   sources: src0 = embedding, src_i = v_all + (i-1)*BT*DIM
//   writes h (bf16) and per-row softmax weights (for deterministic reduce).
// ---------------------------------------------------------------------------
__global__ __launch_bounds__(256)
void mix_kernel(const float* __restrict__ emb,
                const float* __restrict__ vbase,
                const float* __restrict__ wl,          // [512]
                __hip_bfloat16* __restrict__ hout,     // [BT][512] bf16
                float* __restrict__ part,              // [BT][9]
                int n)
{
    const int row = blockIdx.x;
    const int tid = threadIdx.x;
    const int d0  = tid, d1 = tid + 256;

    __shared__ float red[16];   // 8 wave partials x {sumsq, dot}
    __shared__ float lg[8];     // logits per source

    const float w0 = wl[d0], w1 = wl[d1];

    float s0[8], s1[8];
    #pragma unroll
    for (int i = 0; i < 8; ++i) {
        if (i < n) {
            const float* src = (i == 0) ? emb : (vbase + (size_t)(i - 1) * BT * DIM);
            float a = src[(size_t)row * DIM + d0];
            float b = src[(size_t)row * DIM + d1];
            s0[i] = a; s1[i] = b;
            float ss = a * a + b * b;
            float dt = w0 * a + w1 * b;
            // wave32 reduction
            #pragma unroll
            for (int off = 16; off > 0; off >>= 1) {
                ss += __shfl_down(ss, off, 32);
                dt += __shfl_down(dt, off, 32);
            }
            if ((tid & 31) == 0) { red[tid >> 5] = ss; red[8 + (tid >> 5)] = dt; }
            __syncthreads();
            if (tid == 0) {
                float sst = 0.f, dtt = 0.f;
                #pragma unroll
                for (int wv = 0; wv < 8; ++wv) { sst += red[wv]; dtt += red[8 + wv]; }
                // logits = dot(w, x * rsqrt(mean(x^2)+eps)) = (w.x)*rsqrt(...)
                lg[i] = dtt * rsqrtf(sst * (1.0f / (float)DIM) + 1e-8f);
            }
            __syncthreads();
        }
    }

    // softmax over n sources (redundant per-thread; n <= 8 scalars)
    float mx = lg[0];
    #pragma unroll
    for (int i = 1; i < 8; ++i) if (i < n) mx = fmaxf(mx, lg[i]);
    float wgt[8];
    float denom = 0.f;
    #pragma unroll
    for (int i = 0; i < 8; ++i) {
        wgt[i] = (i < n) ? __expf(lg[i] - mx) : 0.0f;
        denom += wgt[i];
    }
    const float inv = 1.0f / denom;

    float h0 = 0.f, h1 = 0.f;
    #pragma unroll
    for (int i = 0; i < 8; ++i) {
        if (i < n) {
            float g = wgt[i] * inv;
            h0 += g * s0[i];
            h1 += g * s1[i];
        }
    }

    // per-row softmax weights for the deterministic reduction (9 slots, pad 0)
    #pragma unroll
    for (int i = 0; i < 9; ++i) {
        if (tid == i) {
            float g = (i < n) ? wgt[i] * inv : 0.0f;
            part[(size_t)row * 9 + i] = g;
        }
    }

    hout[(size_t)row * DIM + d0] = __float2bfloat16(h0);
    hout[(size_t)row * DIM + d1] = __float2bfloat16(h1);
}

// ---------------------------------------------------------------------------
// reduce_w: deterministic mean over BT rows for one layer's 9 weight slots
// ---------------------------------------------------------------------------
__global__ __launch_bounds__(256)
void reduce_w_kernel(const float* __restrict__ part, float* __restrict__ wsum_l) {
    __shared__ float sm[256];
    const int i = blockIdx.x;      // 0..8
    float s = 0.f;
    for (int r = threadIdx.x; r < BT; r += 256) s += part[(size_t)r * 9 + i];
    sm[threadIdx.x] = s;
    __syncthreads();
    for (int off = 128; off > 0; off >>= 1) {
        if ((int)threadIdx.x < off) sm[threadIdx.x] += sm[threadIdx.x + off];
        __syncthreads();
    }
    if (threadIdx.x == 0) wsum_l[i] = sm[0] * (1.0f / (float)BT);
}

// ---------------------------------------------------------------------------
// gemm_kernel: D[M=16384][512] f32 = A[M][512] bf16 @ W (via Wt bf16 [N][K])
//   128x128 block tile, 8 waves (2x4), each wave 64x32 (4x2 WMMA frags),
//   K-step 32, LDS-staged tiles, v_wmma_f32_16x16x32_bf16 inner loop.
// ---------------------------------------------------------------------------
__global__ __launch_bounds__(256)
void gemm_kernel(const __hip_bfloat16* __restrict__ A,   // [M][512] bf16 row-major
                 const __hip_bfloat16* __restrict__ Bt,  // [512(N)][512(K)] bf16
                 float* __restrict__ Dm)                 // [M][512] f32
{
    __shared__ __align__(16) unsigned char lA[128 * 64]; // 128 rows x 32 k x 2B
    __shared__ __align__(16) unsigned char lB[128 * 64]; // 128 n-rows x 32 k x 2B

    const int tid  = threadIdx.x;
    const int wave = tid >> 5;
    const int lane = tid & 31;
    const int wm   = wave >> 2;        // 0..1  (M direction, 64 rows each)
    const int wn   = wave & 3;         // 0..3  (N direction, 32 cols each)
    const int m0   = blockIdx.x * 128;
    const int n0   = blockIdx.y * 128;
    const int half = lane >> 4;        // which 16-lane half
    const int l16  = lane & 15;

    v8f acc[4][2];
    #pragma unroll
    for (int mi = 0; mi < 4; ++mi)
        #pragma unroll
        for (int ni = 0; ni < 2; ++ni)
            acc[mi][ni] = (v8f)(0.0f);

    const char* Ab = (const char*)A;
    const char* Bb = (const char*)Bt;

    for (int k0 = 0; k0 < DIM; k0 += 32) {
        // ---- stage 128x32 bf16 tiles of A and B into LDS (b128 each) ----
        #pragma unroll
        for (int j = 0; j < 2; ++j) {
            int flat = tid * 2 + j;          // 0..511
            int r = flat >> 2;               // 0..127
            int c = flat & 3;                // 16B chunk within 64B row
            *(uint4*)(lA + r * 64 + c * 16) =
                *(const uint4*)(Ab + (size_t)(m0 + r) * (DIM * 2) + k0 * 2 + c * 16);
            *(uint4*)(lB + r * 64 + c * 16) =
                *(const uint4*)(Bb + (size_t)(n0 + r) * (DIM * 2) + k0 * 2 + c * 16);
        }
        // prefetch next K tile while we compute (global_prefetch_b8)
        if (k0 + 32 < DIM) {
            __builtin_prefetch(Ab + (size_t)(m0 + (tid >> 1)) * (DIM * 2) +
                               (k0 + 32) * 2 + (tid & 1) * 32, 0, 1);
            __builtin_prefetch(Bb + (size_t)(n0 + (tid >> 1)) * (DIM * 2) +
                               (k0 + 32) * 2 + (tid & 1) * 32, 0, 1);
        }
        __syncthreads();

        // ---- load fragments per ISA 16-bit A/B wave32 layouts ----
        FragBF16 afr[4], bfr[2];
        #pragma unroll
        for (int mi = 0; mi < 4; ++mi) {
            int rowm = wm * 64 + mi * 16 + l16;
            // elems 0..7: K = half*8 + 0..7 ; elems 8..15: K = 16 + half*8 + 0..7
            afr[mi].u[0] = *(const uint4*)(lA + rowm * 64 + half * 16);
            afr[mi].u[1] = *(const uint4*)(lA + rowm * 64 + 32 + half * 16);
        }
        #pragma unroll
        for (int ni = 0; ni < 2; ++ni) {
            int rown = wn * 32 + ni * 16 + l16;
            // elems 0..7: K = half*8 + 0..7 ; elems 8..15: K = 16 + half*8 + 0..7 (column N = l16)
            bfr[ni].u[0] = *(const uint4*)(lB + rown * 64 + half * 16);
            bfr[ni].u[1] = *(const uint4*)(lB + rown * 64 + 32 + half * 16);
        }

        // ---- 8 WMMAs per wave per K-step ----
        #pragma unroll
        for (int mi = 0; mi < 4; ++mi)
            #pragma unroll
            for (int ni = 0; ni < 2; ++ni)
                acc[mi][ni] = __builtin_amdgcn_wmma_f32_16x16x32_bf16(
                    false, afr[mi].v, false, bfr[ni].v,
                    (short)0, acc[mi][ni], false, false);

        __syncthreads();
    }

    // ---- epilogue: C/D layout -> global f32 ----
    #pragma unroll
    for (int mi = 0; mi < 4; ++mi) {
        #pragma unroll
        for (int ni = 0; ni < 2; ++ni) {
            int col = n0 + wn * 32 + ni * 16 + l16;
            #pragma unroll
            for (int r = 0; r < 8; ++r) {
                int rowm = m0 + wm * 64 + mi * 16 + r + half * 8;
                Dm[(size_t)rowm * DIM + col] = acc[mi][ni][r];
            }
        }
    }
}

// ---------------------------------------------------------------------------
// finalize: apply lower-triangular mask into d_out tail
// ---------------------------------------------------------------------------
__global__ void finalize_kernel(const float* __restrict__ wsum, float* __restrict__ outw) {
    int i = threadIdx.x;
    if (i < NLAYERS * (NLAYERS + 1)) {
        int rowl = i / (NLAYERS + 1);
        int col  = i % (NLAYERS + 1);
        outw[i] = (col <= rowl) ? wsum[i] : 0.0f;
    }
}

// ---------------------------------------------------------------------------
extern "C" void kernel_launch(void* const* d_in, const int* in_sizes, int n_in,
                              void* d_out, int out_size, void* d_ws, size_t ws_size,
                              hipStream_t stream) {
    (void)in_sizes; (void)n_in; (void)out_size; (void)ws_size;

    const float* emb = (const float*)d_in[0];   // [8,2048,512]
    const float* w   = (const float*)d_in[1];   // [8,512]
    const float* lw  = (const float*)d_in[2];   // [8,512,512]

    // workspace layout
    char* ws = (char*)d_ws;
    float* v_all = (float*)ws;                                   // 7 x [BT][512] f32
    size_t off = (size_t)7 * BT * DIM * sizeof(float);
    __hip_bfloat16* h_buf = (__hip_bfloat16*)(ws + off);         // [BT][512] bf16
    off += (size_t)BT * DIM * sizeof(__hip_bfloat16);
    __hip_bfloat16* wt = (__hip_bfloat16*)(ws + off);            // [8][N][K] bf16
    off += (size_t)NLAYERS * DIM * DIM * sizeof(__hip_bfloat16);
    float* part = (float*)(ws + off);                            // [BT][9]
    off += (size_t)BT * 9 * sizeof(float);
    float* wsum = (float*)(ws + off);                            // [8][9]

    float* outv = (float*)d_out;                                 // [8,2048,512]
    float* outw = outv + (size_t)BT * DIM;                       // [8,9]

    // one-time weight transpose+convert (deterministic, re-run each call)
    prep_w_kernel<<<(NLAYERS * DIM * DIM) / 256, 256, 0, stream>>>(lw, wt);

    for (int l = 0; l < NLAYERS; ++l) {
        mix_kernel<<<BT, 256, 0, stream>>>(emb, v_all, w + (size_t)l * DIM,
                                           h_buf, part, l + 1);
        reduce_w_kernel<<<9, 256, 0, stream>>>(part, wsum + (size_t)l * 9);
        float* vout = (l == NLAYERS - 1) ? outv : (v_all + (size_t)l * BT * DIM);
        gemm_kernel<<<dim3(BT / 128, DIM / 128), 256, 0, stream>>>(
            h_buf, wt + (size_t)l * DIM * DIM, vout);
    }

    finalize_kernel<<<1, 128, 0, stream>>>(wsum, outw);
}
